// SentenceLSTM_91053306675501
// MI455X (gfx1250) — compile-verified, matrix operand order checked by reading, not run on hardware
//
#include <hip/hip_runtime.h>
#include <math.h>

// ---------------------------------------------------------------------------
// Problem constants (from setup_inputs)
// ---------------------------------------------------------------------------
#define BATCH   512
#define NVIS    49          // 7x7 grid
#define VDIM    768
#define SEMN    10
#define SEMD    256
#define HID     512
#define SENTIN  1024
#define WORD    512
#define STOP    256
#define TSTEPS  12

typedef __attribute__((ext_vector_type(8)))  __bf16 v8bf;
typedef __attribute__((ext_vector_type(16))) __bf16 v16bf;
typedef __attribute__((ext_vector_type(8)))  float  v8f;

// ---------------------------------------------------------------------------
// Fragment loaders.  Per CDNA5 ISA 16-bit A/B layout: lane<16 holds
// K {kt..kt+7, kt+16..kt+23}; lane>=16 holds K {kt+8..kt+15, kt+24..kt+31}
// (the +8 is folded into the per-lane base pointer).
// ---------------------------------------------------------------------------
__device__ __forceinline__ v16bf load_frag_bf16(const __bf16* __restrict__ p) {
    v8bf lo = *(const v8bf*)(p);
    v8bf hi = *(const v8bf*)(p + 16);
    return __builtin_shufflevector(lo, hi, 0,1,2,3,4,5,6,7,8,9,10,11,12,13,14,15);
}

__device__ __forceinline__ v16bf load_frag_f32(const float* __restrict__ p) {
    v16bf a;
#pragma unroll
    for (int i = 0; i < 8; ++i) {
        a[i]     = (__bf16)p[i];
        a[i + 8] = (__bf16)p[16 + i];
    }
    return a;
}

// ---------------------------------------------------------------------------
// One-wave 16x64 tile accumulation (4 x 16x16 WMMA per 32-K step, A reused).
// All four B fragments are loaded into distinct registers BEFORE the WMMA
// burst so the loads issue as one clause and overlap with matrix math.
// XF32 selects the A-operand element type at compile time.
// ---------------------------------------------------------------------------
template <bool XF32>
__device__ __forceinline__ void wmma_acc4(v8f acc[4],
                                          const void* __restrict__ X, int ldx,
                                          const __bf16* __restrict__ W, int ldw,
                                          int m0, int n0, int K) {
    const int lane = threadIdx.x & 31;
    const int sub  = lane & 15;
    const int half = lane >> 4;
    const size_t arow = (size_t)(m0 + sub) * ldx + half * 8;
    const __bf16* wp = W + (size_t)(n0 + sub) * ldw + half * 8;
#pragma unroll 2
    for (int kt = 0; kt < K; kt += 32) {
        v16bf a;
        if (XF32) a = load_frag_f32((const float*)X + arow + kt);
        else      a = load_frag_bf16((const __bf16*)X + arow + kt);
        v16bf bfrag[4];
#pragma unroll
        for (int j = 0; j < 4; ++j)
            bfrag[j] = load_frag_bf16(wp + (size_t)(j * 16) * ldw + kt);
#pragma unroll
        for (int j = 0; j < 4; ++j)
            acc[j] = __builtin_amdgcn_wmma_f32_16x16x32_bf16(
                /*neg_a=*/false, a, /*neg_b=*/false, bfrag[j],
                /*c_mod=*/(short)0, acc[j], /*reuse_a=*/false, /*reuse_b=*/false);
    }
}

// ---------------------------------------------------------------------------
// Generic dual GEMM: D = X1*W1^T (+ X2*W2^T) (+ b1) (+ b2), optional tanh.
// Writes f32 D (ld ldd) if non-null and/or bf16 Dbf (ld N) if non-null.
// Grid: x = N/256 (4 waves/block, 16x64 tile per wave), y = M/16. Block 128.
// ---------------------------------------------------------------------------
template <bool X1F32>
__global__ void gemm_dual_wmma(const void* __restrict__ X1,
                               const __bf16* __restrict__ W1, int K1,
                               const __bf16* __restrict__ X2,
                               const __bf16* __restrict__ W2, int K2,
                               const float* __restrict__ b1, const float* __restrict__ b2,
                               float* __restrict__ D, long ldd,
                               __bf16* __restrict__ Dbf,
                               int N, int do_tanh) {
    const int wave = threadIdx.x >> 5;
    const int n0 = (blockIdx.x * 4 + wave) * 64;
    const int m0 = blockIdx.y * 16;
    if (n0 >= N) return;

    v8f acc[4];
#pragma unroll
    for (int j = 0; j < 4; ++j) acc[j] = (v8f){0.f,0.f,0.f,0.f,0.f,0.f,0.f,0.f};

    wmma_acc4<X1F32>(acc, X1, K1, W1, K1, m0, n0, K1);
    if (X2) wmma_acc4<false>(acc, X2, K2, W2, K2, m0, n0, K2);

    const int lane = threadIdx.x & 31;
    const int sub  = lane & 15;
    const int half = lane >> 4;
#pragma unroll
    for (int j = 0; j < 4; ++j) {
        const int n = n0 + j * 16 + sub;
        float bias = 0.f;
        if (b1) bias += b1[n];
        if (b2) bias += b2[n];
#pragma unroll
        for (int r = 0; r < 8; ++r) {
            float v = acc[j][r] + bias;
            if (do_tanh) v = tanhf(v);
            const size_t row = (size_t)(m0 + r + 8 * half);
            if (D)   D[row * ldd + n] = v;
            if (Dbf) Dbf[row * (size_t)N + n] = (__bf16)v;
        }
    }
}

// ---------------------------------------------------------------------------
// In-place BatchNorm (eval) + L2 row normalization (f32).
// chmode 0: channel = column.   chmode 1: channel = row % C.
// ---------------------------------------------------------------------------
__global__ void bn_l2n_kernel(float* __restrict__ x,
                              const float* __restrict__ g, const float* __restrict__ b,
                              const float* __restrict__ rm, const float* __restrict__ rv,
                              int D, int chmode, int C) {
    __shared__ float buf[768];
    __shared__ float red[256];
    const long row = blockIdx.x;
    float* xr = x + row * (long)D;
    float ss = 0.f;
    if (chmode == 1) {
        const int ch = (int)(row % C);
        const float sc = g[ch] / sqrtf(rv[ch] + 1e-5f);
        const float sh = b[ch] - rm[ch] * sc;
        for (int d = threadIdx.x; d < D; d += blockDim.x) {
            float v = xr[d] * sc + sh;
            buf[d] = v; ss += v * v;
        }
    } else {
        for (int d = threadIdx.x; d < D; d += blockDim.x) {
            const float sc = g[d] / sqrtf(rv[d] + 1e-5f);
            float v = (xr[d] - rm[d]) * sc + b[d];
            buf[d] = v; ss += v * v;
        }
    }
    red[threadIdx.x] = ss; __syncthreads();
    for (int s = 128; s > 0; s >>= 1) {
        if ((int)threadIdx.x < s) red[threadIdx.x] += red[threadIdx.x + s];
        __syncthreads();
    }
    const float inv = 1.f / (sqrtf(red[0]) + 1e-12f);
    for (int d = threadIdx.x; d < D; d += blockDim.x) xr[d] = buf[d] * inv;
}

// ---------------------------------------------------------------------------
// Co-attention: S[n] = 1 - l2nX[b,n,:]·q[b,:]; relu; l2n over n; softmax(0.3*);
// ctx_in[b, out_off+d] = sum_n alpha[n]*feats[b,n,d]  (written as bf16).
// One block (256 thr = 8 waves) per batch element; wave-parallel dots.
// ---------------------------------------------------------------------------
__global__ void coatt_kernel(const float* __restrict__ l2nX, const float* __restrict__ feats,
                             const float* __restrict__ q,   __bf16* __restrict__ ctx_in,
                             int N, int D, int out_off) {
    __shared__ float S[64];
    __shared__ float alpha[64];
    const int b    = blockIdx.x;
    const int wave = threadIdx.x >> 5;
    const int lane = threadIdx.x & 31;
    const float* qr = q + (long)b * D;

    for (int n = wave; n < N; n += 8) {
        const float* xr = l2nX + ((long)b * N + n) * D;
        float p = 0.f;
        for (int d = lane; d < D; d += 32) p += xr[d] * qr[d];
#pragma unroll
        for (int o = 16; o > 0; o >>= 1) p += __shfl_xor(p, o, 32);
        if (lane == 0) S[n] = 1.f - p;
    }
    __syncthreads();
    if (threadIdx.x == 0) {
        float nrm = 0.f;
        for (int n = 0; n < N; ++n) { float r = fmaxf(S[n], 0.f); S[n] = r; nrm += r * r; }
        const float inv = 0.3f / (sqrtf(nrm) + 1e-12f);
        float mx = -1e30f;
        for (int n = 0; n < N; ++n) { S[n] *= inv; mx = fmaxf(mx, S[n]); }
        float se = 0.f;
        for (int n = 0; n < N; ++n) { float e = expf(S[n] - mx); alpha[n] = e; se += e; }
        const float rs = 1.f / se;
        for (int n = 0; n < N; ++n) alpha[n] *= rs;
    }
    __syncthreads();
    for (int d = threadIdx.x; d < D; d += blockDim.x) {
        float acc = 0.f;
        for (int n = 0; n < N; ++n) acc += alpha[n] * feats[((long)b * N + n) * D + d];
        ctx_in[(long)b * SENTIN + out_off + d] = (__bf16)acc;
    }
}

// ---------------------------------------------------------------------------
// LSTM pointwise (torch gate order i,f,g,o). c in place (f32); h_new -> bf16.
// ---------------------------------------------------------------------------
__global__ void lstm_pointwise(const float* __restrict__ gates, float* __restrict__ c,
                               __bf16* __restrict__ hnew_bf) {
    const int idx = blockIdx.x * blockDim.x + threadIdx.x;
    if (idx >= BATCH * HID) return;
    const int b = idx / HID, j = idx % HID;
    const float* gr = gates + (long)b * 4 * HID;
    const float gi = gr[j], gf = gr[HID + j], gg = gr[2 * HID + j], go = gr[3 * HID + j];
    const float si = 1.f / (1.f + expf(-gi));
    const float sf = 1.f / (1.f + expf(-gf));
    const float so = 1.f / (1.f + expf(-go));
    const float cn = sf * c[idx] + si * tanhf(gg);
    c[idx] = cn;
    hnew_bf[idx] = (__bf16)(so * tanhf(cn));
}

// ---------------------------------------------------------------------------
// Stop head: ps[b,t,:] = p0[b,:] @ fs_w^T + fs_b   (N=2, K=256)
// ---------------------------------------------------------------------------
__global__ void fs_kernel(const float* __restrict__ p0, const float* __restrict__ fw,
                          const float* __restrict__ fb, float* __restrict__ out_ps, int t) {
    const int idx = blockIdx.x * blockDim.x + threadIdx.x;
    if (idx >= BATCH * 2) return;
    const int b = idx >> 1, j = idx & 1;
    const float* pr = p0 + (long)b * STOP;
    const float* wr = fw + (long)j * STOP;
    float acc = fb[j];
    for (int k = 0; k < STOP; ++k) acc += pr[k] * wr[k];
    out_ps[(long)b * TSTEPS * 2 + t * 2 + j] = acc;
}

__global__ void cvt_f32_to_bf16(const float* __restrict__ s, __bf16* __restrict__ d, int n) {
    const int i = blockIdx.x * blockDim.x + threadIdx.x;
    if (i < n) d[i] = (__bf16)s[i];
}

__global__ void fill_zero_f32(float* __restrict__ p, int n) {
    const int i = blockIdx.x * blockDim.x + threadIdx.x;
    if (i < n) p[i] = 0.f;
}
__global__ void fill_zero_bf16(__bf16* __restrict__ p, int n) {
    const int i = blockIdx.x * blockDim.x + threadIdx.x;
    if (i < n) p[i] = (__bf16)0.f;
}

// ---------------------------------------------------------------------------
extern "C" void kernel_launch(void* const* d_in, const int* in_sizes, int n_in,
                              void* d_out, int out_size, void* d_ws, size_t ws_size,
                              hipStream_t stream) {
    const float* vis    = (const float*)d_in[0];   // (B,7,7,768) == (B*49,768)
    const float* sem    = (const float*)d_in[1];   // (B,10,256)
    const float* Wv_w   = (const float*)d_in[4];
    const float* Wv_b   = (const float*)d_in[5];
    const float* bnv_g  = (const float*)d_in[6];
    const float* bnv_b  = (const float*)d_in[7];
    const float* bnv_rm = (const float*)d_in[8];
    const float* bnv_rv = (const float*)d_in[9];
    const float* Wvh_w  = (const float*)d_in[10];
    const float* Wvh_b  = (const float*)d_in[11];
    const float* bvh_g  = (const float*)d_in[12];
    const float* bvh_b  = (const float*)d_in[13];
    const float* bvh_rm = (const float*)d_in[14];
    const float* bvh_rv = (const float*)d_in[15];
    const float* Wa_w   = (const float*)d_in[16];
    const float* Wa_b   = (const float*)d_in[17];
    const float* bna_g  = (const float*)d_in[18];
    const float* bna_b  = (const float*)d_in[19];
    const float* bna_rm = (const float*)d_in[20];
    const float* bna_rv = (const float*)d_in[21];
    const float* Wah_w  = (const float*)d_in[22];
    const float* Wah_b  = (const float*)d_in[23];
    const float* bah_g  = (const float*)d_in[24];
    const float* bah_b  = (const float*)d_in[25];
    const float* bah_rm = (const float*)d_in[26];
    const float* bah_rv = (const float*)d_in[27];
    const float* Wfc_w  = (const float*)d_in[28];
    const float* Wfc_b  = (const float*)d_in[29];
    const float* Wih    = (const float*)d_in[30];
    const float* Whh    = (const float*)d_in[31];
    const float* bih    = (const float*)d_in[32];
    const float* bhh    = (const float*)d_in[33];
    const float* th_w   = (const float*)d_in[34];
    const float* th_b   = (const float*)d_in[35];
    const float* tc_w   = (const float*)d_in[36];
    const float* tc_b   = (const float*)d_in[37];
    const float* sp_w   = (const float*)d_in[38];
    const float* sp_b   = (const float*)d_in[39];
    const float* sc_w   = (const float*)d_in[40];
    const float* sc_b   = (const float*)d_in[41];
    const float* fs_w   = (const float*)d_in[42];
    const float* fs_b   = (const float*)d_in[43];

    float* out_topics = (float*)d_out;                                   // (B,T,512)
    float* out_ps     = (float*)d_out + (size_t)BATCH * TSTEPS * WORD;   // (B,T,2)

    // ---- workspace (byte bump allocator, 256B aligned chunks) ----
    char* wsb = (char*)d_ws;
    size_t off = 0;
    auto alloc = [&](size_t bytes) -> void* {
        void* p = wsb + off;
        off += (bytes + 255) & ~(size_t)255;
        return p;
    };
    float*  wv     = (float*)alloc((size_t)BATCH * NVIS * VDIM * 4);  // l2n(bn(vis@Wv^T))
    float*  wa     = (float*)alloc((size_t)BATCH * SEMN * SEMD * 4);
    float*  wvh    = (float*)alloc((size_t)BATCH * VDIM * 4);
    float*  wah    = (float*)alloc((size_t)BATCH * SEMD * 4);
    float*  gates  = (float*)alloc((size_t)BATCH * 4 * HID * 4);
    float*  cst    = (float*)alloc((size_t)BATCH * HID * 4);
    float*  p0     = (float*)alloc((size_t)BATCH * STOP * 4);
    __bf16* ctxin  = (__bf16*)alloc((size_t)BATCH * SENTIN * 2);
    __bf16* ctx_bf = (__bf16*)alloc((size_t)BATCH * SENTIN * 2);
    __bf16* hA     = (__bf16*)alloc((size_t)BATCH * HID * 2);
    __bf16* hB     = (__bf16*)alloc((size_t)BATCH * HID * 2);
    // bf16 weight copies (loop-invariant, L2-resident)
    __bf16* Wv_bf  = (__bf16*)alloc((size_t)VDIM * VDIM * 2);
    __bf16* Wvh_bf = (__bf16*)alloc((size_t)VDIM * HID * 2);
    __bf16* Wa_bf  = (__bf16*)alloc((size_t)SEMD * SEMD * 2);
    __bf16* Wah_bf = (__bf16*)alloc((size_t)SEMD * HID * 2);
    __bf16* Wfc_bf = (__bf16*)alloc((size_t)SENTIN * SENTIN * 2);
    __bf16* Wih_bf = (__bf16*)alloc((size_t)4 * HID * SENTIN * 2);
    __bf16* Whh_bf = (__bf16*)alloc((size_t)4 * HID * HID * 2);
    __bf16* th_bf  = (__bf16*)alloc((size_t)WORD * HID * 2);
    __bf16* tc_bf  = (__bf16*)alloc((size_t)WORD * SENTIN * 2);
    __bf16* sp_bf  = (__bf16*)alloc((size_t)STOP * HID * 2);
    __bf16* sc_bf  = (__bf16*)alloc((size_t)STOP * HID * 2);

    // ---- one-time weight conversion f32 -> bf16 ----
    auto cvt = [&](const float* s, __bf16* d, int n) {
        cvt_f32_to_bf16<<<(n + 255) / 256, 256, 0, stream>>>(s, d, n);
    };
    cvt(Wv_w,  Wv_bf,  VDIM * VDIM);
    cvt(Wvh_w, Wvh_bf, VDIM * HID);
    cvt(Wa_w,  Wa_bf,  SEMD * SEMD);
    cvt(Wah_w, Wah_bf, SEMD * HID);
    cvt(Wfc_w, Wfc_bf, SENTIN * SENTIN);
    cvt(Wih,   Wih_bf, 4 * HID * SENTIN);
    cvt(Whh,   Whh_bf, 4 * HID * HID);
    cvt(th_w,  th_bf,  WORD * HID);
    cvt(tc_w,  tc_bf,  WORD * SENTIN);
    cvt(sp_w,  sp_bf,  STOP * HID);
    cvt(sc_w,  sc_bf,  STOP * HID);

    // ---- init recurrent state ----
    fill_zero_bf16<<<(BATCH * HID + 255) / 256, 256, 0, stream>>>(hA, BATCH * HID);
    fill_zero_f32 <<<(BATCH * HID + 255) / 256, 256, 0, stream>>>(cst, BATCH * HID);

    // ---- hoisted loop-invariant co-attention projections ----
    gemm_dual_wmma<true><<<dim3(VDIM / 256, (BATCH * NVIS) / 16), 128, 0, stream>>>(
        vis, Wv_bf, VDIM, nullptr, nullptr, 0, Wv_b, nullptr,
        wv, VDIM, nullptr, VDIM, 0);
    bn_l2n_kernel<<<BATCH * NVIS, 256, 0, stream>>>(wv, bnv_g, bnv_b, bnv_rm, bnv_rv,
                                                    VDIM, /*chmode=*/1, NVIS);
    gemm_dual_wmma<true><<<dim3(SEMD / 256, (BATCH * SEMN) / 16), 128, 0, stream>>>(
        sem, Wa_bf, SEMD, nullptr, nullptr, 0, Wa_b, nullptr,
        wa, SEMD, nullptr, SEMD, 0);
    bn_l2n_kernel<<<BATCH * SEMN, 256, 0, stream>>>(wa, bna_g, bna_b, bna_rm, bna_rv,
                                                    SEMD, /*chmode=*/1, SEMN);

    // ---- recurrent loop ----
    for (int t = 0; t < TSTEPS; ++t) {
        __bf16* hprev = (t & 1) ? hB : hA;
        __bf16* hnext = (t & 1) ? hA : hB;

        // wvh = l2n(bn(h @ Wvh^T + b))
        gemm_dual_wmma<false><<<dim3(VDIM / 256, BATCH / 16), 128, 0, stream>>>(
            hprev, Wvh_bf, HID, nullptr, nullptr, 0, Wvh_b, nullptr,
            wvh, VDIM, nullptr, VDIM, 0);
        bn_l2n_kernel<<<BATCH, 256, 0, stream>>>(wvh, bvh_g, bvh_b, bvh_rm, bvh_rv,
                                                 VDIM, 0, VDIM);
        // wah = l2n(bn(h @ Wah^T + b))
        gemm_dual_wmma<false><<<dim3(SEMD / 256, BATCH / 16), 128, 0, stream>>>(
            hprev, Wah_bf, HID, nullptr, nullptr, 0, Wah_b, nullptr,
            wah, SEMD, nullptr, SEMD, 0);
        bn_l2n_kernel<<<BATCH, 256, 0, stream>>>(wah, bah_g, bah_b, bah_rm, bah_rv,
                                                 SEMD, 0, SEMD);

        // attention -> ctxin(bf16) = [v_att | a_att]
        coatt_kernel<<<BATCH, 256, 0, stream>>>(wv, vis, wvh, ctxin, NVIS, VDIM, 0);
        coatt_kernel<<<BATCH, 256, 0, stream>>>(wa, sem, wah, ctxin, SEMN, SEMD, VDIM);

        // ctx(bf16) = ctxin @ Wfc^T + b
        gemm_dual_wmma<false><<<dim3(SENTIN / 256, BATCH / 16), 128, 0, stream>>>(
            ctxin, Wfc_bf, SENTIN, nullptr, nullptr, 0, Wfc_b, nullptr,
            nullptr, 0, ctx_bf, SENTIN, 0);

        // gates(f32) = ctx @ Wih^T + h @ Whh^T + bih + bhh
        gemm_dual_wmma<false><<<dim3((4 * HID) / 256, BATCH / 16), 128, 0, stream>>>(
            ctx_bf, Wih_bf, SENTIN, hprev, Whh_bf, HID, bih, bhh,
            gates, 4 * HID, nullptr, 4 * HID, 0);

        // c,h update (h written as bf16)
        lstm_pointwise<<<(BATCH * HID + 255) / 256, 256, 0, stream>>>(gates, cst, hnext);

        // topic = tanh(h_new @ th^T + ctx @ tc^T + biases) -> out_topics[:, t, :]
        gemm_dual_wmma<false><<<dim3(WORD / 256, BATCH / 16), 128, 0, stream>>>(
            hnext, th_bf, HID, ctx_bf, tc_bf, SENTIN, th_b, tc_b,
            out_topics + (size_t)t * WORD, (long)TSTEPS * WORD, nullptr, WORD, 1);

        // p0 = tanh(h_prev @ sp^T + h_new @ sc^T + biases)
        gemm_dual_wmma<false><<<dim3(STOP / 256, BATCH / 16), 128, 0, stream>>>(
            hprev, sp_bf, HID, hnext, sc_bf, HID, sp_b, sc_b,
            p0, STOP, nullptr, STOP, 1);

        // ps[:, t, :] = p0 @ fs_w^T + fs_b
        fs_kernel<<<(BATCH * 2 + 255) / 256, 256, 0, stream>>>(p0, fs_w, fs_b, out_ps, t);
    }
}
